// DeepSAT_36275293782333
// MI455X (gfx1250) — compile-verified, matrix operand order checked by reading, not run on hardware
//
#include <hip/hip_runtime.h>
#include <hip/hip_bf16.h>
#include <math.h>

typedef __attribute__((ext_vector_type(16))) _Float16 v16h;
typedef __attribute__((ext_vector_type(8)))  float    v8f;

#define NN 200000
#define NE 400000
#define DD 128
#define DG 384
#define DM 256
#define NL 20

// ---------------- WMMA helpers (CDNA5 wave32, 16x16x32 f16 -> f32) ----------

__device__ __forceinline__ v8f wmma32(v16h a, v16h b, v8f c) {
  return __builtin_amdgcn_wmma_f32_16x16x32_f16(false, a, false, b, (short)0, c,
                                                false, false);
}

// A fragment: 16 rows x 32 K, source is fp32 row pointer (row = lane&15 chosen
// by caller). elements 0..7 = K koff..koff+7, 8..15 = K 16+koff..16+koff+7.
__device__ __forceinline__ v16h load_a_f32(const float* __restrict__ row,
                                           int kb, int koff) {
  v16h a;
  const float* p0 = row + kb + koff;
  const float* p1 = row + kb + 16 + koff;
#pragma unroll
  for (int e = 0; e < 8; ++e) {
    a[e]     = (_Float16)p0[e];
    a[8 + e] = (_Float16)p1[e];
  }
  return a;
}

// B fragment for X @ W^T: lane's column o -> row o of torch-layout weight
// (contiguous in K).  wrow = W16 + o*Kd.
__device__ __forceinline__ v16h load_b_w16(const _Float16* __restrict__ wrow,
                                           int kb, int koff) {
  v16h b;
  const _Float16* q = wrow + kb + koff;
#pragma unroll
  for (int e = 0; e < 8; ++e) {
    b[e]     = q[e];
    b[8 + e] = q[16 + e];
  }
  return b;
}

// ---------------- setup kernels --------------------------------------------

__global__ void k_cvt(const float* __restrict__ s, _Float16* __restrict__ d, int n) {
  int i = blockIdx.x * 256 + threadIdx.x;
  if (i < n) d[i] = (_Float16)s[i];
}

__global__ void k_init_h(float* __restrict__ h, const float* __restrict__ ew,
                         const float* __restrict__ eb) {
  long i = (long)blockIdx.x * 256 + threadIdx.x;
  if (i >= (long)NN * DD) return;
  int d = (int)(i & (DD - 1));
  h[i] = ew[d] + eb[d];
}

__global__ void k_zero_lv(int* a, int* b) {
  int t = threadIdx.x;
  if (t < NL) { a[t] = 0; b[t] = 0; }
}

__global__ void k_count(const int* __restrict__ fl, int* cnt) {
  int n = blockIdx.x * 256 + threadIdx.x;
  if (n < NN) atomicAdd(&cnt[fl[n]], 1);
}

__global__ void k_scan(const int* __restrict__ cnt, int* offs) {
  if (threadIdx.x == 0) {
    int a = 0;
    for (int i = 0; i < NL; ++i) { offs[i] = a; a += cnt[i]; }
  }
}

__global__ void k_fill(const int* __restrict__ fl, const int* __restrict__ offs,
                       int* cur, int* list) {
  int n = blockIdx.x * 256 + threadIdx.x;
  if (n >= NN) return;
  int l = fl[n];
  int pos = offs[l] + atomicAdd(&cur[l], 1);
  list[pos] = n;
}

// zero msg rows only for nodes at this level (one float4 per thread)
__global__ void k_zero_msg(float* __restrict__ msg, const int* __restrict__ fl,
                           int lvl) {
  long t = (long)blockIdx.x * 256 + threadIdx.x;
  if (t >= (long)NN * 32) return;
  int node = (int)(t >> 5);
  if (fl[node] != lvl) return;
  ((float4*)msg)[t] = make_float4(0.f, 0.f, 0.f, 0.f);
}

// ---------------- per-level GEMM: h_lin = h @ aggr_w^T + aggr_b ------------
// block = 256 threads = 8 waves, each wave: 16 rows x 128 cols.

__global__ void k_gemm_hlin(const float* __restrict__ X,
                            const _Float16* __restrict__ W,
                            const float* __restrict__ bias,
                            float* __restrict__ Y) {
  int wave = threadIdx.x >> 5, lane = threadIdx.x & 31;
  long rowbase = ((long)blockIdx.x * 8 + wave) * 16;
  if (rowbase >= NN) return;
  int mn = lane & 15, hi = lane >> 4, koff = hi * 8;
  const float* xr = X + (rowbase + mn) * DD;
  v16h A[4];
#pragma unroll
  for (int kc = 0; kc < 4; ++kc) A[kc] = load_a_f32(xr, kc * 32, koff);
#pragma unroll
  for (int nt = 0; nt < 8; ++nt) {
    int o = nt * 16 + mn;
    const _Float16* wr = W + (size_t)o * DD;
    v8f acc = {};
#pragma unroll
    for (int kc = 0; kc < 4; ++kc)
      acc = wmma32(A[kc], load_b_w16(wr, kc * 32, koff), acc);
    float bv = bias[o];
#pragma unroll
    for (int v = 0; v < 8; ++v)
      Y[(rowbase + v + 8 * hi) * DD + o] = acc[v] + bv;
  }
}

// ---------------- scatter: msg[dst] += h_lin[src] for dst at level l -------

__global__ void k_scatter(const int* __restrict__ eidx, const int* __restrict__ fl,
                          const float* __restrict__ hlin, float* __restrict__ msg,
                          int lvl) {
  int e = blockIdx.x * 256 + threadIdx.x;
  if (e >= NE) return;
  int d = eidx[NE + e];
  if (fl[d] != lvl) return;
  int s = eidx[e];
  const float* hs = hlin + (size_t)s * DD;
  float* md = msg + (size_t)d * DD;
#pragma unroll 4
  for (int c = 0; c < DD; ++c) atomicAdd(&md[c], hs[c]);
}

// ---------------- GRU update for compacted level-l nodes -------------------
// block = 128 threads = 4 waves sharing one 16-row tile; gi/gh staged in LDS.

__global__ void k_gru(const float* __restrict__ msg, float* __restrict__ h,
                      const _Float16* __restrict__ wih16,
                      const _Float16* __restrict__ whh16,
                      const float* __restrict__ bih, const float* __restrict__ bhh,
                      const int* __restrict__ list, const int* __restrict__ offs,
                      const int* __restrict__ cnts, int lvl) {
  __shared__ float GI[16][DG];
  __shared__ float GH[16][DG];
  int count = cnts[lvl];
  int off   = offs[lvl];
  int rowbase = blockIdx.x * 16;
  if (rowbase >= count) return;

  int wave = threadIdx.x >> 5, lane = threadIdx.x & 31;
  int mn = lane & 15, hi = lane >> 4, koff = hi * 8;
  int ridx = rowbase + mn;
  if (ridx >= count) ridx = count - 1;      // clamp (results unused)
  int node = list[off + ridx];

  v16h Am[4], Ah[4];
  const float* mr = msg + (size_t)node * DD;
  const float* hr = h + (size_t)node * DD;
#pragma unroll
  for (int kc = 0; kc < 4; ++kc) {
    Am[kc] = load_a_f32(mr, kc * 32, koff);
    Ah[kc] = load_a_f32(hr, kc * 32, koff);
  }

  // 48 output tiles: 24 for gi (msg@wih^T), 24 for gh (h@whh^T)
  for (int t = wave; t < 48; t += 4) {
    int isH = (t >= 24);
    int c = isH ? (t - 24) : t;
    int o = c * 16 + mn;
    const _Float16* wr = (isH ? whh16 : wih16) + (size_t)o * DD;
    v8f acc = {};
#pragma unroll
    for (int kc = 0; kc < 4; ++kc)
      acc = wmma32(isH ? Ah[kc] : Am[kc], load_b_w16(wr, kc * 32, koff), acc);
    float bv = (isH ? bhh : bih)[o];
    float* dst = isH ? &GH[0][0] : &GI[0][0];
#pragma unroll
    for (int v = 0; v < 8; ++v)
      dst[(v + 8 * hi) * DG + o] = acc[v] + bv;
  }
  __syncthreads();

  // elementwise GRU over 16x128 elements
  for (int idx = threadIdx.x; idx < 16 * DD; idx += 128) {
    int m = idx >> 7, cc = idx & 127;
    if (rowbase + m >= count) continue;
    int nd = list[off + rowbase + m];
    float ir = GI[m][cc], iz = GI[m][DD + cc], inn = GI[m][2 * DD + cc];
    float hrv = GH[m][cc], hz = GH[m][DD + cc], hn = GH[m][2 * DD + cc];
    float r = 1.f / (1.f + __expf(-(ir + hrv)));
    float z = 1.f / (1.f + __expf(-(iz + hz)));
    float nnv = tanhf(inn + r * hn);
    size_t p = (size_t)nd * DD + cc;
    float ho = h[p];
    h[p] = (1.f - z) * nnv + z * ho;
  }
}

// ---------------- fused MLP head: relu(relu(h W1^T+b1) W2^T+b2) W3^T+b3 ----
// block = 128 = 4 waves, one 16-row tile per wave; z1 staged f16 in LDS,
// z2 fused directly into w3 dot with cross-lane reduction.

__global__ void k_mlp(const float* __restrict__ h,
                      const _Float16* __restrict__ w1_16, const float* __restrict__ b1,
                      const _Float16* __restrict__ w2_16, const float* __restrict__ b2,
                      const float* __restrict__ w3, const float* __restrict__ b3,
                      float* __restrict__ out) {
  __shared__ _Float16 Z1[4][16][DM];     // 32 KB
  int wave = threadIdx.x >> 5, lane = threadIdx.x & 31;
  long rowbase = ((long)blockIdx.x * 4 + wave) * 16;
  if (rowbase >= NN) return;
  int mn = lane & 15, hi = lane >> 4, koff = hi * 8;

  const float* xr = h + (rowbase + mn) * DD;
  v16h A[4];
#pragma unroll
  for (int kc = 0; kc < 4; ++kc) A[kc] = load_a_f32(xr, kc * 32, koff);

  _Float16* z1w = &Z1[wave][0][0];
  // GEMM1: 128 -> 256, relu, store f16 to LDS
#pragma unroll
  for (int nt = 0; nt < 16; ++nt) {
    int o = nt * 16 + mn;
    const _Float16* wr = w1_16 + (size_t)o * DD;
    v8f acc = {};
#pragma unroll
    for (int kc = 0; kc < 4; ++kc)
      acc = wmma32(A[kc], load_b_w16(wr, kc * 32, koff), acc);
    float bv = b1[o];
#pragma unroll
    for (int v = 0; v < 8; ++v)
      z1w[(v + 8 * hi) * DM + o] = (_Float16)fmaxf(acc[v] + bv, 0.f);
  }

  // A fragments for GEMM2 gathered from LDS (intra-wave, DScnt-ordered)
  v16h A2[8];
#pragma unroll
  for (int kc = 0; kc < 8; ++kc) {
    int kb = kc * 32;
    const _Float16* zr = z1w + mn * DM;
#pragma unroll
    for (int e = 0; e < 8; ++e) {
      A2[kc][e]     = zr[kb + koff + e];
      A2[kc][8 + e] = zr[kb + 16 + koff + e];
    }
  }

  // GEMM2 (256->256) fused with relu and w3 dot product
  float part[8];
#pragma unroll
  for (int v = 0; v < 8; ++v) part[v] = 0.f;
#pragma unroll
  for (int nt = 0; nt < 16; ++nt) {
    int o = nt * 16 + mn;
    const _Float16* wr = w2_16 + (size_t)o * DM;
    v8f acc = {};
#pragma unroll
    for (int kc = 0; kc < 8; ++kc)
      acc = wmma32(A2[kc], load_b_w16(wr, kc * 32, koff), acc);
    float bv = b2[o], w3v = w3[o];
#pragma unroll
    for (int v = 0; v < 8; ++v)
      part[v] += fmaxf(acc[v] + bv, 0.f) * w3v;
  }
  // reduce over the 16 lanes holding different output columns
#pragma unroll
  for (int mask = 1; mask <= 8; mask <<= 1)
#pragma unroll
    for (int v = 0; v < 8; ++v)
      part[v] += __shfl_xor(part[v], mask, 32);
  if (mn == 0) {
#pragma unroll
    for (int v = 0; v < 8; ++v)
      out[rowbase + v + 8 * hi] = part[v] + b3[0];
  }
}

// ---------------- host driver ----------------------------------------------

extern "C" void kernel_launch(void* const* d_in, const int* in_sizes, int n_in,
                              void* d_out, int out_size, void* d_ws, size_t ws_size,
                              hipStream_t stream) {
  const int*   fl     = (const int*)d_in[0];
  const int*   eidx   = (const int*)d_in[1];
  const float* emd_w  = (const float*)d_in[2];
  const float* emd_b  = (const float*)d_in[3];
  const float* aggr_w = (const float*)d_in[4];
  const float* aggr_b = (const float*)d_in[5];
  const float* wih    = (const float*)d_in[6];
  const float* whh    = (const float*)d_in[7];
  const float* bih    = (const float*)d_in[8];
  const float* bhh    = (const float*)d_in[9];
  const float* w1     = (const float*)d_in[10];
  const float* b1     = (const float*)d_in[11];
  const float* w2     = (const float*)d_in[12];
  const float* b2     = (const float*)d_in[13];
  const float* w3     = (const float*)d_in[14];
  const float* b3     = (const float*)d_in[15];
  float* out = (float*)d_out;
  (void)in_sizes; (void)n_in; (void)out_size; (void)ws_size;

  char* ws = (char*)d_ws;
  size_t off = 0;
  auto alloc = [&](size_t bytes) -> char* {
    off = (off + 255) & ~(size_t)255;
    char* p = ws + off;
    off += bytes;
    return p;
  };
  float*    h      = (float*)alloc((size_t)NN * DD * 4);
  float*    hlin   = (float*)alloc((size_t)NN * DD * 4);
  float*    msg    = (float*)alloc((size_t)NN * DD * 4);
  _Float16* aggr16 = (_Float16*)alloc((size_t)DD * DD * 2);
  _Float16* wih16  = (_Float16*)alloc((size_t)DG * DD * 2);
  _Float16* whh16  = (_Float16*)alloc((size_t)DG * DD * 2);
  _Float16* w1_16  = (_Float16*)alloc((size_t)DM * DD * 2);
  _Float16* w2_16  = (_Float16*)alloc((size_t)DM * DM * 2);
  int* lcnt = (int*)alloc(NL * 4);
  int* lcur = (int*)alloc(NL * 4);
  int* loff = (int*)alloc(NL * 4);
  int* list = (int*)alloc((size_t)NN * 4);

  // one-time (per launch) weight conversion + init + level compaction
  k_cvt<<<(DD * DD + 255) / 256, 256, 0, stream>>>(aggr_w, aggr16, DD * DD);
  k_cvt<<<(DG * DD + 255) / 256, 256, 0, stream>>>(wih, wih16, DG * DD);
  k_cvt<<<(DG * DD + 255) / 256, 256, 0, stream>>>(whh, whh16, DG * DD);
  k_cvt<<<(DM * DD + 255) / 256, 256, 0, stream>>>(w1, w1_16, DM * DD);
  k_cvt<<<(DM * DM + 255) / 256, 256, 0, stream>>>(w2, w2_16, DM * DM);
  k_init_h<<<(NN * DD + 255) / 256, 256, 0, stream>>>(h, emd_w, emd_b);
  k_zero_lv<<<1, 64, 0, stream>>>(lcnt, lcur);
  k_count<<<(NN + 255) / 256, 256, 0, stream>>>(fl, lcnt);
  k_scan<<<1, 32, 0, stream>>>(lcnt, loff);
  k_fill<<<(NN + 255) / 256, 256, 0, stream>>>(fl, loff, lcur, list);

  for (int l = 1; l < NL; ++l) {
    k_zero_msg<<<(NN * 32 + 255) / 256, 256, 0, stream>>>(msg, fl, l);
    k_gemm_hlin<<<(NN + 127) / 128, 256, 0, stream>>>(h, aggr16, aggr_b, hlin);
    k_scatter<<<(NE + 255) / 256, 256, 0, stream>>>(eidx, fl, hlin, msg, l);
    k_gru<<<(NN + 15) / 16, 128, 0, stream>>>(msg, h, wih16, whh16, bih, bhh,
                                              list, loff, lcnt, l);
  }

  k_mlp<<<(NN / 16 + 3) / 4, 128, 0, stream>>>(h, w1_16, b1, w2_16, b2, w3, b3, out);
}
